// StandardAttention_14018773254729
// MI455X (gfx1250) — matrix operand order repaired and checked by reading, not compiled, not run on hardware
//
#include <hip/hip_runtime.h>

// ---------------------------------------------------------------------------
// Types and helpers
// ---------------------------------------------------------------------------
typedef __bf16 bf16x16 __attribute__((ext_vector_type(16)));
typedef float  v8f     __attribute__((ext_vector_type(8)));
typedef unsigned int v4u __attribute__((ext_vector_type(4)));
typedef int          v4i __attribute__((ext_vector_type(4)));
typedef int          v8i __attribute__((ext_vector_type(8)));

union FragU { bf16x16 v; unsigned int u[8]; };

__device__ __forceinline__ unsigned short f2bf(float f) {
  unsigned int x = __float_as_uint(f);
  x += 0x7FFFu + ((x >> 16) & 1u);   // round-to-nearest-even
  return (unsigned short)(x >> 16);
}

__device__ __forceinline__ v8f v8f_zero() {
  v8f r;
  #pragma unroll
  for (int i = 0; i < 8; ++i) r[i] = 0.0f;
  return r;
}

// ---------------------------------------------------------------------------
// Tensor Data Mover: 2D bf16 tile (tile_x cols * tile_y rows, row stride
// `stride_elems`) from global -> LDS, packed contiguously at lds_off.
// D# layout per CDNA5 ISA.
// ---------------------------------------------------------------------------
__device__ __forceinline__ void tdm_load_2d_bf16(unsigned lds_off,
                                                 const void* gptr,
                                                 unsigned tile_x, unsigned tile_y,
                                                 unsigned stride_elems) {
  unsigned long long ga = (unsigned long long)gptr;
  v4u g0;
  g0[0] = 1u;                                          // count=1, user mode
  g0[1] = lds_off;                                     // lds_addr (bytes)
  g0[2] = (unsigned)(ga & 0xFFFFFFFFu);                // global_addr[31:0]
  g0[3] = (unsigned)((ga >> 32) & 0x1FFFFFFu)          // global_addr[56:32]
          | (2u << 30);                                // type = 2 ("image")
  unsigned td0 = tile_x, td1 = tile_y;                 // tensor dims == tile
  unsigned long long st0 = stride_elems;               // dim0 stride (elems)
  v8i g1;
  g1[0] = (int)(1u << 16);                             // data_size=1 (2B), mask=0
  g1[1] = (int)((td0 & 0xFFFFu) << 16);                // tensor_dim0[15:0]
  g1[2] = (int)((td0 >> 16) | ((td1 & 0xFFFFu) << 16));// dim0[31:16], dim1[15:0]
  g1[3] = (int)((td1 >> 16) | (tile_x << 16));         // dim1[31:16], tile_dim0
  g1[4] = (int)(tile_y & 0xFFFFu);                     // tile_dim1 (tile_dim2=0)
  g1[5] = (int)(st0 & 0xFFFFFFFFu);                    // stride0[31:0]
  g1[6] = (int)((st0 >> 32) & 0xFFFFu);                // stride0[47:32], stride1=0
  g1[7] = 0;
  v4i gz = {0, 0, 0, 0};
#if __clang_major__ >= 23
  v8i gz8 = {0, 0, 0, 0, 0, 0, 0, 0};
  __builtin_amdgcn_tensor_load_to_lds(g0, g1, gz, gz, gz8, 0);
#else
  __builtin_amdgcn_tensor_load_to_lds(g0, g1, gz, gz, 0);
#endif
}

#define NB   4
#define NS   2048
#define NH   16
#define ND   128
#define HID  2048
#define NQKV 6144

// ---------------------------------------------------------------------------
// fp32 -> bf16 conversion
// ---------------------------------------------------------------------------
__global__ void convert_f32_bf16(const float* __restrict__ src,
                                 unsigned short* __restrict__ dst,
                                 long long n) {
  long long i = (long long)blockIdx.x * blockDim.x + threadIdx.x;
  if (i < n) dst[i] = f2bf(src[i]);
}

// ---------------------------------------------------------------------------
// bf16 GEMM: C[M,N] (fp32) = A[M,K] (bf16) @ B[K,N] (bf16)
// Block tile 128x128x64; 8 waves in 4x2, each wave owns 32x64 (2x4 WMMA grid).
// A tile staged via the Tensor Data Mover; B tile staged transposed manually.
// ---------------------------------------------------------------------------
__global__ __launch_bounds__(256) void gemm_bf16_f32(
    const unsigned short* __restrict__ A,
    const unsigned short* __restrict__ Bm,
    float* __restrict__ C, int M, int N, int K) {
  __shared__ unsigned short As[128][64];   // [m][k]  (TDM destination, packed)
  __shared__ unsigned short Bs[128][64];   // [n][k]  (transposed for pair loads)

  const int tid  = threadIdx.x;
  const int wave = tid >> 5, lane = tid & 31;
  const int hx   = lane >> 4, l16 = lane & 15;
  const int wm   = wave >> 1, wn = wave & 1;
  const int m0   = blockIdx.y * 128, n0 = blockIdx.x * 128;
  const unsigned as_off = (unsigned)(unsigned long long)(void*)&As[0][0];

  v8f acc[2][4];
  #pragma unroll
  for (int i = 0; i < 2; ++i)
    #pragma unroll
    for (int j = 0; j < 4; ++j) acc[i][j] = v8f_zero();

  for (int kt = 0; kt < K; kt += 64) {
    // A tile (128 x 64, row stride K) via TDM, issued by wave 0.
    if (wave == 0)
      tdm_load_2d_bf16(as_off, A + (size_t)m0 * K + kt, 64, 128, (unsigned)K);

    // B tile staged transposed: 8192 elems, 8 per vec load, 32 per thread.
    #pragma unroll
    for (int i = 0; i < 4; ++i) {
      int idx = (i * 256 + tid) * 8;           // element index, n fastest
      int bn = idx & 127, bk = idx >> 7;
      uint4 w = *(const uint4*)&Bm[(size_t)(kt + bk) * N + n0 + bn];
      unsigned wv[4] = {w.x, w.y, w.z, w.w};
      #pragma unroll
      for (int j = 0; j < 4; ++j) {
        Bs[bn + 2 * j][bk]     = (unsigned short)(wv[j] & 0xFFFFu);
        Bs[bn + 2 * j + 1][bk] = (unsigned short)(wv[j] >> 16);
      }
    }
    if (wave == 0) __builtin_amdgcn_s_wait_tensorcnt(0);
    __syncthreads();

    #pragma unroll
    for (int ks = 0; ks < 2; ++ks) {
      const int ko = ks * 32;
      FragU a[2], b;
      #pragma unroll
      for (int mt = 0; mt < 2; ++mt)
        #pragma unroll
        for (int v = 0; v < 8; ++v) {
          int kb = (v >> 2) * 16 + hx * 8 + (v & 3) * 2;   // A-matrix K pair
          a[mt].u[v] = *(const unsigned int*)&As[wm * 32 + mt * 16 + l16][ko + kb];
        }
      #pragma unroll
      for (int nt = 0; nt < 4; ++nt) {
        #pragma unroll
        for (int v = 0; v < 8; ++v) {
          int kb = (v >> 2) * 16 + hx * 8 + (v & 3) * 2;   // B-matrix K pair
          b.u[v] = *(const unsigned int*)&Bs[wn * 64 + nt * 16 + l16][ko + kb];
        }
        #pragma unroll
        for (int mt = 0; mt < 2; ++mt)
          acc[mt][nt] = __builtin_amdgcn_wmma_f32_16x16x32_bf16(
              false, a[mt].v, false, b.v, (short)0, acc[mt][nt], false, false);
      }
    }
    __syncthreads();
  }

  #pragma unroll
  for (int mt = 0; mt < 2; ++mt)
    #pragma unroll
    for (int nt = 0; nt < 4; ++nt) {
      float* cp = C + (size_t)(m0 + wm * 32 + mt * 16 + 8 * hx) * N
                    + (n0 + wn * 64 + nt * 16 + l16);
      #pragma unroll
      for (int v = 0; v < 8; ++v) cp[(size_t)v * N] = acc[mt][nt][v];
    }
}

// ---------------------------------------------------------------------------
// RoPE + precision demotion. qkv fp32 [B*S][6144] -> Q,K,V bf16 [B*S][2048].
// Score scale D^-0.5 folded into Q. One thread per (b,s,h,d<64) pair.
// ---------------------------------------------------------------------------
__global__ void rope_kernel(const float* __restrict__ qkv,
                            const float* __restrict__ cosb,
                            const float* __restrict__ sinb,
                            unsigned short* __restrict__ Qb,
                            unsigned short* __restrict__ Kb,
                            unsigned short* __restrict__ Vb) {
  int idx = blockIdx.x * blockDim.x + threadIdx.x;   // [0, B*S*H*64)
  int d = idx & 63;
  int h = (idx >> 6) & (NH - 1);
  int s = (idx >> 10) & (NS - 1);
  int b = idx >> 21;

  size_t row = (size_t)b * NS + s;
  const float* base = qkv + row * NQKV;
  int c0i = h * ND + d, c1i = c0i + 64;

  float c0 = cosb[s * ND + d],      c1 = cosb[s * ND + d + 64];
  float s0 = sinb[s * ND + d],      s1 = sinb[s * ND + d + 64];
  float q0 = base[c0i],             q1 = base[c1i];
  float k0 = base[HID + c0i],       k1 = base[HID + c1i];
  float v0 = base[2 * HID + c0i],   v1 = base[2 * HID + c1i];

  float qo0 = q0 * c0 - q1 * s0, qo1 = q1 * c1 + q0 * s1;
  float ko0 = k0 * c0 - k1 * s0, ko1 = k1 * c1 + k0 * s1;
  const float scale = 0.08838834764831845f;   // 128^-0.5

  size_t ob = row * HID + c0i;
  Qb[ob] = f2bf(qo0 * scale);  Qb[ob + 64] = f2bf(qo1 * scale);
  Kb[ob] = f2bf(ko0);          Kb[ob + 64] = f2bf(ko1);
  Vb[ob] = f2bf(v0);           Vb[ob + 64] = f2bf(v1);
}

// ---------------------------------------------------------------------------
// Flash attention. One workgroup per (qblock=128 rows, head, batch);
// 8 waves, each owning 16 q rows x D=128. Iterates k in chunks of 32.
// K chunk staged via TDM (row-major); V chunk staged transposed manually.
// ---------------------------------------------------------------------------
__global__ __launch_bounds__(256) void flash_kernel(
    const unsigned short* __restrict__ Qb,
    const unsigned short* __restrict__ Kb,
    const unsigned short* __restrict__ Vb,
    unsigned short* __restrict__ Ob) {
  __shared__ unsigned short Ks[32][128];     // [k][d]  (TDM destination)
  __shared__ unsigned short Vt[128][32];     // [d][k]  (transposed)
  __shared__ unsigned short Ps[8][16][32];   // per-wave P scratch [m][k]

  const int qblk = blockIdx.x, h = blockIdx.y, b = blockIdx.z;
  const int tid  = threadIdx.x;
  const int wave = tid >> 5, lane = tid & 31;
  const int hx   = lane >> 4, l16 = lane & 15;
  const size_t srow0 = (size_t)b * NS;
  const int qr0 = qblk * 128 + wave * 16;
  const unsigned ks_off = (unsigned)(unsigned long long)(void*)&Ks[0][0];

  // Q strip (16 x 128) held in registers as 4 A-fragments (K chunks of 32).
  unsigned int aq[4][8];
  {
    const unsigned short* qrow = Qb + (srow0 + qr0 + l16) * HID + h * ND;
    #pragma unroll
    for (int c = 0; c < 4; ++c)
      #pragma unroll
      for (int v = 0; v < 8; ++v) {
        int kb = c * 32 + (v >> 2) * 16 + hx * 8 + (v & 3) * 2;
        aq[c][v] = *(const unsigned int*)(qrow + kb);
      }
  }

  v8f accO[8];
  #pragma unroll
  for (int i = 0; i < 8; ++i) accO[i] = v8f_zero();
  float mr[8], lr[8];
  #pragma unroll
  for (int i = 0; i < 8; ++i) { mr[i] = -3.0e38f; lr[i] = 0.0f; }

  for (int kc = 0; kc < NS / 32; ++kc) {
    // K chunk (32 x 128, row stride HID) via TDM, issued by wave 0.
    if (wave == 0)
      tdm_load_2d_bf16(ks_off, Kb + (srow0 + kc * 32) * HID + h * ND,
                       128, 32, HID);

    // V chunk staged transposed: 4096 elems, 8 per vec load, 16 per thread.
    #pragma unroll
    for (int i = 0; i < 2; ++i) {
      int idx = (i * 256 + tid) * 8;          // element index, d fastest
      int d = idx & 127, kk = idx >> 7;
      uint4 w = *(const uint4*)&Vb[(srow0 + kc * 32 + kk) * HID + h * ND + d];
      unsigned wv[4] = {w.x, w.y, w.z, w.w};
      #pragma unroll
      for (int j = 0; j < 4; ++j) {
        Vt[d + 2 * j][kk]     = (unsigned short)(wv[j] & 0xFFFFu);
        Vt[d + 2 * j + 1][kk] = (unsigned short)(wv[j] >> 16);
      }
    }
    if (kc + 1 < NS / 32) {   // hint next chunk into cache (global_prefetch_b8)
      size_t g = (srow0 + (kc + 1) * 32 + (tid >> 3)) * HID + h * ND + (tid & 7) * 16;
      __builtin_prefetch(Kb + g, 0, 0);
      __builtin_prefetch(Vb + g, 0, 0);
    }
    if (wave == 0) __builtin_amdgcn_s_wait_tensorcnt(0);
    __syncthreads();

    // Scores: S (16 x 32) = Qstrip @ Ks^T, two 16x16 WMMA tiles, K=128.
    v8f accS[2];
    accS[0] = v8f_zero(); accS[1] = v8f_zero();
    FragU af, bf;
    #pragma unroll
    for (int nh = 0; nh < 2; ++nh)
      #pragma unroll
      for (int c = 0; c < 4; ++c) {
        #pragma unroll
        for (int v = 0; v < 8; ++v) {
          int kb = (v >> 2) * 16 + hx * 8 + (v & 3) * 2;
          bf.u[v] = *(const unsigned int*)&Ks[nh * 16 + l16][c * 32 + kb];
        }
        #pragma unroll
        for (int v = 0; v < 8; ++v) af.u[v] = aq[c][v];
        accS[nh] = __builtin_amdgcn_wmma_f32_16x16x32_bf16(
            false, af.v, false, bf.v, (short)0, accS[nh], false, false);
      }

    // Online softmax per row (row m = v + 8*hx; 16-lane reductions per half).
    #pragma unroll
    for (int v = 0; v < 8; ++v) {
      float x0 = accS[0][v], x1 = accS[1][v];
      float loc = fmaxf(x0, x1);
      #pragma unroll
      for (int msk = 1; msk < 16; msk <<= 1)
        loc = fmaxf(loc, __shfl_xor(loc, msk, 32));
      float mn   = fmaxf(mr[v], loc);
      float corr = __expf(mr[v] - mn);
      float p0 = __expf(x0 - mn), p1 = __expf(x1 - mn);
      float rs = p0 + p1;
      #pragma unroll
      for (int msk = 1; msk < 16; msk <<= 1)
        rs += __shfl_xor(rs, msk, 32);
      lr[v] = lr[v] * corr + rs;
      mr[v] = mn;
      #pragma unroll
      for (int dt = 0; dt < 8; ++dt) accO[dt][v] *= corr;
      int rowm = v + 8 * hx;
      Ps[wave][rowm][l16]      = f2bf(p0);
      Ps[wave][rowm][16 + l16] = f2bf(p1);
    }
    asm volatile("s_wait_dscnt 0" ::: "memory");   // wave-local LDS visibility

    // Reload P (16 x 32) in A-fragment layout, then O += P @ Vchunk.
    #pragma unroll
    for (int v = 0; v < 8; ++v) {
      int kb = (v >> 2) * 16 + hx * 8 + (v & 3) * 2;
      af.u[v] = *(const unsigned int*)&Ps[wave][l16][kb];
    }
    #pragma unroll
    for (int dt = 0; dt < 8; ++dt) {
      #pragma unroll
      for (int v = 0; v < 8; ++v) {
        int kb = (v >> 2) * 16 + hx * 8 + (v & 3) * 2;
        bf.u[v] = *(const unsigned int*)&Vt[dt * 16 + l16][kb];
      }
      accO[dt] = __builtin_amdgcn_wmma_f32_16x16x32_bf16(
          false, af.v, false, bf.v, (short)0, accO[dt], false, false);
    }
    __syncthreads();
  }

  // Epilogue: normalize by row sum, write bf16 [B*S][H*D].
  unsigned short* op = Ob + (srow0 + qr0 + 8 * hx) * HID + h * ND + l16;
  #pragma unroll
  for (int dt = 0; dt < 8; ++dt)
    #pragma unroll
    for (int v = 0; v < 8; ++v)
      op[(size_t)v * HID + dt * 16] = f2bf(accO[dt][v] / lr[v]);
}

// ---------------------------------------------------------------------------
// Launcher
// ---------------------------------------------------------------------------
extern "C" void kernel_launch(void* const* d_in, const int* in_sizes, int n_in,
                              void* d_out, int out_size, void* d_ws, size_t ws_size,
                              hipStream_t stream) {
  const float* cos_t = (const float*)d_in[0];
  const float* sin_t = (const float*)d_in[1];
  const float* hs    = (const float*)d_in[2];
  const float* Wqkv  = (const float*)d_in[3];
  const float* Wo    = (const float*)d_in[4];
  float* out = (float*)d_out;

  const size_t ROWS = (size_t)NB * NS;                 // 8192
  char* ws = (char*)d_ws;
  size_t off = 0;
  unsigned short* Xb    = (unsigned short*)(ws + off); off += ROWS * HID * 2;
  unsigned short* Wqkvb = (unsigned short*)(ws + off); off += (size_t)HID * NQKV * 2;
  unsigned short* Wob   = (unsigned short*)(ws + off); off += (size_t)HID * HID * 2;
  float*          qkv   = (float*)(ws + off);
  unsigned short* AttnO = (unsigned short*)(ws + off); // reuse qkv region
  off += ROWS * NQKV * 4;
  unsigned short* Qb = (unsigned short*)(ws + off); off += ROWS * HID * 2;
  unsigned short* Kb = (unsigned short*)(ws + off); off += ROWS * HID * 2;
  unsigned short* Vb = (unsigned short*)(ws + off); off += ROWS * HID * 2;

  // 1. fp32 -> bf16 conversions
  {
    long long n = (long long)ROWS * HID;
    convert_f32_bf16<<<(unsigned)((n + 255) / 256), 256, 0, stream>>>(hs, Xb, n);
    n = (long long)HID * NQKV;
    convert_f32_bf16<<<(unsigned)((n + 255) / 256), 256, 0, stream>>>(Wqkv, Wqkvb, n);
    n = (long long)HID * HID;
    convert_f32_bf16<<<(unsigned)((n + 255) / 256), 256, 0, stream>>>(Wo, Wob, n);
  }

  // 2. QKV projection: [8192,2048] @ [2048,6144] -> fp32
  gemm_bf16_f32<<<dim3(NQKV / 128, ROWS / 128), 256, 0, stream>>>(
      Xb, Wqkvb, qkv, (int)ROWS, NQKV, HID);

  // 3. RoPE + demote to bf16 (folds D^-0.5 into Q)
  {
    long long n = (long long)NB * NS * NH * 64;
    rope_kernel<<<(unsigned)(n / 256), 256, 0, stream>>>(qkv, cos_t, sin_t, Qb, Kb, Vb);
  }

  // 4. Flash attention -> AttnO bf16 [8192][2048]
  flash_kernel<<<dim3(NS / 128, NH, NB), 256, 0, stream>>>(Qb, Kb, Vb, AttnO);

  // 5. Output projection: [8192,2048] @ [2048,2048] -> d_out fp32
  gemm_bf16_f32<<<dim3(HID / 128, ROWS / 128), 256, 0, stream>>>(
      AttnO, Wob, out, (int)ROWS, HID, HID);
}
